// HybridMemoryBlock_44547400794262
// MI455X (gfx1250) — matrix operand order repaired and checked by reading, not compiled, not run on hardware
//
#include <hip/hip_runtime.h>
#include <hip/hip_bf16.h>
#include <math.h>

// ---------------- types ----------------
typedef _Float16 h16;
typedef __attribute__((ext_vector_type(16))) _Float16 v16h;
typedef __attribute__((ext_vector_type(8)))  _Float16 v8h;
typedef __attribute__((ext_vector_type(8)))  float    v8f;

#define WMMA_F32_F16(a, b, c) \
  __builtin_amdgcn_wmma_f32_16x16x32_f16(false, (a), false, (b), (short)0, (c), false, false)

// Problem constants
#define BB 2
#define LL 2048
#define DD 256
#define PP 32
#define MM (BB * LL)  // 4096

// ---------------- WMMA operand loaders (NT convention) ----------------
// A operand (16x32 f16): lane m=lane&15 holds row m; halves 0..7 -> k=kc+base..+7,
// halves 8..15 -> k=kc+16+base..+7 (base = (lane>=16)?8:0). Two contiguous v8h loads.
__device__ inline v16h load_a16(const h16* p, int row, int ld, int kc, int hi8) {
  const h16* q = p + (size_t)row * ld + kc + hi8;
  v8h lo = *(const v8h*)(q);
  v8h hi = *(const v8h*)(q + 16);
  v16h r;
#pragma unroll
  for (int i = 0; i < 8; ++i) { r[i] = lo[i]; r[i + 8] = hi[i]; }
  return r;
}
// B operand (32x16 f16): lane n=lane&15 holds column n; halves 0..7 -> k=kc+base..+7,
// halves 8..15 -> k=kc+16+base..+7 (base = (lane>=16)?8:0). Two v8h loads from NxK.
__device__ inline v16h load_b16(const h16* p, int col, int ld, int kc, int k16) {
  const h16* q = p + (size_t)col * ld + kc + (k16 >> 1);
  v8h lo = *(const v8h*)(q);
  v8h hi = *(const v8h*)(q + 16);
  v16h r;
#pragma unroll
  for (int i = 0; i < 8; ++i) { r[i] = lo[i]; r[i + 8] = hi[i]; }
  return r;
}

__device__ inline float gelu_f(float v) {
  return 0.5f * v * (1.0f + erff(v * 0.70710678118654752f));
}

// ---------------- gfx1250 async global->LDS copy (ASYNCcnt tracked) -----------------
// GVS mode: mem_addr = SGPR_base + VGPR_offset; LDS dest offset in a VGPR.
__device__ inline void async_copy16(unsigned lds_off, const h16* gbase, int byte_off) {
  asm volatile("global_load_async_to_lds_b128 %0, %1, %2"
               :
               : "v"(lds_off), "v"(byte_off), "s"(gbase)
               : "memory");
}
__device__ inline void async_wait0() {
  asm volatile("s_wait_asynccnt 0x0" ::: "memory");
}

// ---------------- weight transpose + f16 convert: w (KxN f32) -> wt (NxK f16) ---------
__global__ void k_transpose_w(const float* __restrict__ w, h16* __restrict__ wt,
                              int K, int N) {
  int idx = blockIdx.x * 256 + threadIdx.x;
  if (idx >= K * N) return;
  int k = idx / N, n = idx - k * N;
  wt[(size_t)n * K + k] = (h16)w[idx];
}

// ---------------- build xh (f16) and gate-input gih = [x, shift1(x)] (f16) ----------
__global__ void k_build_x(const float* __restrict__ x, h16* __restrict__ xh,
                          h16* __restrict__ gih) {
  int idx = blockIdx.x * 256 + threadIdx.x;      // over MM*DD
  int m = idx >> 8, d = idx & 255;
  float xv = x[idx];
  xh[idx] = (h16)xv;
  gih[(size_t)m * 512 + d] = (h16)xv;
  int l = m & (LL - 1);
  float xs = (l == 0) ? 0.0f : x[idx - DD];
  gih[(size_t)m * 512 + 256 + d] = (h16)xs;
}

// ---------------- positional Q/K features: [cos φ, sin φ] (L x 64 f16) --------------
__global__ void k_qpos(const float* __restrict__ ph, h16* __restrict__ qpos) {
  int idx = blockIdx.x * 256 + threadIdx.x;      // over LL*PP
  int l = idx >> 5, p = idx & 31;
  float t = ph[idx];
  qpos[(size_t)l * 64 + p]      = (h16)cosf(t);
  qpos[(size_t)l * 64 + 32 + p] = (h16)sinf(t);
}

// ---------------- kv Q/K features (gate folded into K, K shifted by 1) --------------
__global__ void k_qkkv(const float* __restrict__ kp, const float* __restrict__ vg,
                       h16* __restrict__ qkv, h16* __restrict__ kkv) {
  int idx = blockIdx.x * 256 + threadIdx.x;      // over MM*PP
  int m = idx >> 5, p = idx & 31;
  float th = kp[idx];
  qkv[(size_t)m * 64 + p]      = (h16)cosf(th);
  qkv[(size_t)m * 64 + 32 + p] = (h16)sinf(th);
  int l = m & (LL - 1);
  float kc = 0.0f, ks = 0.0f;
  if (l > 0) {
    float tp = kp[idx - PP];
    float g = vg[m];
    kc = g * cosf(tp);
    ks = g * sinf(tp);
  }
  kkv[(size_t)m * 64 + p]      = (h16)kc;
  kkv[(size_t)m * 64 + 32 + p] = (h16)ks;
}

// ---------------- generic WMMA GEMM (1 tile/wave), used for N=32 --------------------
// mode 4: tanh*pi -> f32
__global__ __launch_bounds__(256) void k_gemm(const h16* __restrict__ A,
                                              const h16* __restrict__ Wt,
                                              const float* __restrict__ bias,
                                              void* __restrict__ outp,
                                              const float* __restrict__ xadd,
                                              int M, int N, int K, int mode) {
  int lane = threadIdx.x & 31, wid = threadIdx.x >> 5;
  int l15 = lane & 15, k16 = lane & 16, hi8 = k16 >> 1;
  int tile = blockIdx.x * 8 + wid;
  int tilesN = N >> 4;
  int mt = tile / tilesN, nt = tile - mt * tilesN;
  if (mt * 16 >= M) return;

  v8f c;
#pragma unroll
  for (int i = 0; i < 8; ++i) c[i] = 0.0f;

  int arow = mt * 16 + l15;
  int bcol = nt * 16 + l15;
  for (int kc = 0; kc < K; kc += 32) {
    v16h a = load_a16(A, arow, K, kc, hi8);
    v16h b = load_b16(Wt, bcol, K, kc, k16);
    c = WMMA_F32_F16(a, b, c);
  }

  int n = nt * 16 + l15;
  float bv = bias ? bias[n] : 0.0f;
#pragma unroll
  for (int r = 0; r < 8; ++r) {
    int m = mt * 16 + r + hi8;
    float v = c[r] + bv;
    size_t idx = (size_t)m * N + n;
    if (mode == 4) {
      ((float*)outp)[idx] = tanhf(v) * 3.14159265358979323846f;
    } else {
      ((float*)outp)[idx] = v;
    }
  }
}

// ---------------- register-blocked WMMA GEMM: 4 N-tiles per wave (N % 64 == 0) ------
// mode 0: f16  1: gelu->f16  3: f16 transposed Vt[(b*N+n)*L + l]  5: f32  6: f32 x+C
__global__ __launch_bounds__(256) void k_gemm4(const h16* __restrict__ A,
                                               const h16* __restrict__ Wt,
                                               const float* __restrict__ bias,
                                               void* __restrict__ outp,
                                               const float* __restrict__ xadd,
                                               int M, int N, int K, int mode) {
  int lane = threadIdx.x & 31, wid = threadIdx.x >> 5;
  int l15 = lane & 15, k16 = lane & 16, hi8 = k16 >> 1;
  int grp = blockIdx.x * 8 + wid;
  int gN = N >> 6;
  int mt = grp / gN, n4 = grp - mt * gN;
  if (mt * 16 >= M) return;

  v8f c[4];
#pragma unroll
  for (int j = 0; j < 4; ++j)
#pragma unroll
    for (int i = 0; i < 8; ++i) c[j][i] = 0.0f;

  int arow = mt * 16 + l15;
  for (int kc = 0; kc < K; kc += 32) {
    v16h a = load_a16(A, arow, K, kc, hi8);
#pragma unroll
    for (int j = 0; j < 4; ++j) {
      v16h b = load_b16(Wt, n4 * 64 + j * 16 + l15, K, kc, k16);
      c[j] = WMMA_F32_F16(a, b, c[j]);
    }
  }

#pragma unroll
  for (int j = 0; j < 4; ++j) {
    int n = n4 * 64 + j * 16 + l15;
    float bv = bias ? bias[n] : 0.0f;
#pragma unroll
    for (int r = 0; r < 8; ++r) {
      int m = mt * 16 + r + hi8;
      float v = c[j][r] + bv;
      size_t idx = (size_t)m * N + n;
      if (mode == 0) {
        ((h16*)outp)[idx] = (h16)v;
      } else if (mode == 1) {
        ((h16*)outp)[idx] = (h16)gelu_f(v);
      } else if (mode == 3) {
        ((h16*)outp)[(((size_t)(m >> 11) * N + n) << 11) + (m & (LL - 1))] = (h16)v;
      } else if (mode == 5) {
        ((float*)outp)[idx] = v;
      } else {  // mode 6
        ((float*)outp)[idx] = xadd[idx] + v;
      }
    }
  }
}

// ---------------- value-gate head: sigmoid(h @ w_vg2 + b), N=1 ----------------------
__global__ __launch_bounds__(256) void k_vg(const h16* __restrict__ hh,
                                            const float* __restrict__ w,
                                            const float* __restrict__ bb,
                                            float* __restrict__ vg) {
  int lane = threadIdx.x & 31, wid = threadIdx.x >> 5;
  int row = blockIdx.x * 8 + wid;
  float s = 0.0f;
  for (int k = lane; k < DD; k += 32) s += (float)hh[(size_t)row * DD + k] * w[k];
#pragma unroll
  for (int m = 16; m; m >>= 1) s += __shfl_xor(s, m, 32);
  if (lane == 0) vg[row] = 1.0f / (1.0f + expf(-(s + bb[0])));
}

// ---------------- blend head: softmax(bgh @ w_bg2 + b), N=2 -------------------------
__global__ __launch_bounds__(256) void k_blend(const h16* __restrict__ bgh,
                                               const float* __restrict__ w,
                                               const float* __restrict__ bb,
                                               float* __restrict__ blend) {
  int lane = threadIdx.x & 31, wid = threadIdx.x >> 5;
  int row = blockIdx.x * 8 + wid;
  float s0 = 0.0f, s1 = 0.0f;
  for (int k = lane; k < 128; k += 32) {
    float hv = (float)bgh[(size_t)row * 128 + k];
    s0 += hv * w[2 * k];
    s1 += hv * w[2 * k + 1];
  }
#pragma unroll
  for (int m = 16; m; m >>= 1) {
    s0 += __shfl_xor(s0, m, 32);
    s1 += __shfl_xor(s1, m, 32);
  }
  if (lane == 0) {
    float l0 = s0 + bb[0], l1 = s1 + bb[1];
    float mx = fmaxf(l0, l1);
    float e0 = expf(l0 - mx), e1 = expf(l1 - mx);
    float inv = 1.0f / (e0 + e1);
    blend[(size_t)row * 2]     = e0 * inv;
    blend[(size_t)row * 2 + 1] = e1 * inv;
  }
}

// ---------------- inclusive gate cumsum -> kv row scale, per batch ------------------
__global__ void k_scan(const float* __restrict__ vg, float* __restrict__ rowscale) {
  __shared__ float buf[256];
  int b = blockIdx.x, tid = threadIdx.x;
  const float inv_sp = 0.17677669529663689f;  // 1/sqrt(32)
  float offset = 0.0f;
  for (int seg = 0; seg < LL / 256; ++seg) {
    int m = b * LL + seg * 256 + tid;
    buf[tid] = vg[m];
    __syncthreads();
    for (int s = 1; s < 256; s <<= 1) {
      float t = (tid >= s) ? buf[tid - s] : 0.0f;
      __syncthreads();
      buf[tid] += t;
      __syncthreads();
    }
    float cum = buf[tid] + offset;
    rowscale[m] = rsqrtf(fmaxf(cum, 1.0f)) * inv_sp;
    float tot = buf[255];
    __syncthreads();
    offset += tot;
  }
}

// ---------------- causal linear attention: O[l,d] = sum_{l'<=l} (Q_l.K_l') V[l',d] ---
// Q,K: (per batch) L x 64 f16 row-major. Vt: per batch D x L f16 (transposed).
// Workgroup: 64 rows x 256 cols of O; 8 waves; chunk loop of 64 columns (l').
// K chunk (8KB) and V chunk (32KB) are staged in LDS via async global->LDS copies.
__global__ __launch_bounds__(256) void k_attn(const h16* __restrict__ Qf,
                                              const h16* __restrict__ Kf,
                                              const h16* __restrict__ Vt,
                                              const float* __restrict__ rowscale,
                                              float cscale, float* __restrict__ outp,
                                              int qkStride) {
  __shared__ alignas(16) h16 Kbuf[64][72];
  __shared__ alignas(16) h16 Vbuf[256][72];
  __shared__ alignas(16) h16 Sbuf[64][72];

  int tid = threadIdx.x;
  int lane = tid & 31, wid = tid >> 5;
  int l15 = lane & 15, k16 = lane & 16, hi8 = k16 >> 1;
  int b = blockIdx.y;
  int rb = blockIdx.x;
  int rowbase = rb * 64;
  const h16* Qb = Qf + (size_t)b * qkStride;
  const h16* Kb = Kf + (size_t)b * qkStride;
  const h16* Vb = Vt + (size_t)b * DD * LL;
  float* Ob = outp + (size_t)b * LL * DD;

  unsigned kb_off = (unsigned)(size_t)&Kbuf[0][0];
  unsigned vb_off = (unsigned)(size_t)&Vbuf[0][0];

  int it = wid & 3;     // row tile 0..3
  int side = wid >> 2;  // column half 0/1 (128 cols each)
  int arow = rowbase + it * 16 + l15;

  v16h aq0 = load_a16(Qb, arow, 64, 0, hi8);
  v16h aq1 = load_a16(Qb, arow, 64, 32, hi8);

  v8f oacc[8];
#pragma unroll
  for (int j = 0; j < 8; ++j)
#pragma unroll
    for (int r = 0; r < 8; ++r) oacc[j][r] = 0.0f;

  for (int ch = 0; ch <= rb; ++ch) {
    int cb = ch * 64;
    // ---- async-stage K chunk (64x64 halves) and V chunk (256x64 halves) to LDS ----
#pragma unroll
    for (int t = 0; t < 2; ++t) {
      int c = tid + t * 256;            // 512 x 16B pieces
      int row = c >> 3, c8 = c & 7;
      async_copy16(kb_off + row * 144 + c8 * 16, Kb,
                   ((cb + row) * 64 + c8 * 8) * 2);
    }
#pragma unroll
    for (int t = 0; t < 8; ++t) {
      int c = tid + t * 256;            // 2048 x 16B pieces
      int d = c >> 3, c8 = c & 7;
      async_copy16(vb_off + d * 144 + c8 * 16, Vb,
                   (d * LL + cb + c8 * 8) * 2);
    }
    async_wait0();
    __syncthreads();

    // ---- scores S = Q @ K^T for this 64x64 chunk (each wave: 2 tiles) ----
#pragma unroll
    for (int jj = 0; jj < 2; ++jj) {
      int j = side * 2 + jj;
      v16h b0 = load_b16(&Kbuf[0][0], j * 16 + l15, 72, 0, k16);
      v16h b1 = load_b16(&Kbuf[0][0], j * 16 + l15, 72, 32, k16);
      v8f s;
#pragma unroll
      for (int r = 0; r < 8; ++r) s[r] = 0.0f;
      s = WMMA_F32_F16(aq0, b0, s);
      s = WMMA_F32_F16(aq1, b1, s);
      if (ch == rb) {  // causal mask on the diagonal chunk (keep l' <= l)
#pragma unroll
        for (int r = 0; r < 8; ++r) {
          int rg = rowbase + it * 16 + r + hi8;
          int cg = cb + j * 16 + l15;
          s[r] = (cg > rg) ? 0.0f : s[r];
        }
      }
#pragma unroll
      for (int r = 0; r < 8; ++r)
        Sbuf[it * 16 + r + hi8][j * 16 + l15] = (h16)s[r];
    }
    __syncthreads();

    // ---- O += S @ V_chunk (both operands from LDS) ----
    v16h as0 = load_a16(&Sbuf[0][0], it * 16 + l15, 72, 0, hi8);
    v16h as1 = load_a16(&Sbuf[0][0], it * 16 + l15, 72, 32, hi8);
#pragma unroll
    for (int jd = 0; jd < 8; ++jd) {
      int d = side * 128 + jd * 16 + l15;
      v16h bv0 = load_b16(&Vbuf[0][0], d, 72, 0, k16);
      v16h bv1 = load_b16(&Vbuf[0][0], d, 72, 32, k16);
      oacc[jd] = WMMA_F32_F16(as0, bv0, oacc[jd]);
      oacc[jd] = WMMA_F32_F16(as1, bv1, oacc[jd]);
    }
    __syncthreads();
  }

#pragma unroll
  for (int jd = 0; jd < 8; ++jd)
#pragma unroll
    for (int r = 0; r < 8; ++r) {
      int l = rowbase + it * 16 + r + hi8;
      int d = side * 128 + jd * 16 + l15;
      float sc = rowscale ? rowscale[b * LL + l] : cscale;
      Ob[(size_t)l * DD + d] = oacc[jd][r] * sc;
    }
}

// ---------------- blended = p0 * pos_ret + p1 * kv_ret ------------------------------
__global__ void k_combine(const float* __restrict__ pr, const float* __restrict__ kr,
                          const float* __restrict__ blend, float* __restrict__ out) {
  int idx = blockIdx.x * 256 + threadIdx.x;  // over MM*DD
  int m = idx >> 8;
  out[idx] = blend[2 * m] * pr[idx] + blend[2 * m + 1] * kr[idx];
}

// ---------------- LayerNorm over D=256, write f16 -----------------------------------
__global__ __launch_bounds__(256) void k_ln(const float* __restrict__ in,
                                            const float* __restrict__ gamma,
                                            const float* __restrict__ beta,
                                            h16* __restrict__ outh) {
  __shared__ float red[256];
  int row = blockIdx.x, tid = threadIdx.x;
  float v = in[(size_t)row * DD + tid];
  red[tid] = v;
  __syncthreads();
  for (int s = 128; s > 0; s >>= 1) {
    if (tid < s) red[tid] += red[tid + s];
    __syncthreads();
  }
  float mu = red[0] * (1.0f / 256.0f);
  __syncthreads();
  float d0 = v - mu;
  red[tid] = d0 * d0;
  __syncthreads();
  for (int s = 128; s > 0; s >>= 1) {
    if (tid < s) red[tid] += red[tid + s];
    __syncthreads();
  }
  float var = red[0] * (1.0f / 256.0f);
  float y = d0 * rsqrtf(var + 1e-5f) * gamma[tid] + beta[tid];
  outh[(size_t)row * DD + tid] = (h16)y;
}

// =====================================================================================
extern "C" void kernel_launch(void* const* d_in, const int* in_sizes, int n_in,
                              void* d_out, int out_size, void* d_ws, size_t ws_size,
                              hipStream_t stream) {
  const float* x        = (const float*)d_in[0];
  const float* pos_ph   = (const float*)d_in[1];
  const float* w_pv     = (const float*)d_in[2];
  const float* b_pv     = (const float*)d_in[3];
  const float* w_key    = (const float*)d_in[4];
  const float* b_key    = (const float*)d_in[5];
  const float* w_kv     = (const float*)d_in[6];
  const float* b_kv     = (const float*)d_in[7];
  const float* w_vg1    = (const float*)d_in[8];
  const float* b_vg1    = (const float*)d_in[9];
  const float* w_vg2    = (const float*)d_in[10];
  const float* b_vg2    = (const float*)d_in[11];
  const float* w_bg1    = (const float*)d_in[12];
  const float* b_bg1    = (const float*)d_in[13];
  const float* w_bg2    = (const float*)d_in[14];
  const float* b_bg2    = (const float*)d_in[15];
  const float* ln_r_g   = (const float*)d_in[16];
  const float* ln_r_b   = (const float*)d_in[17];
  const float* w_r1     = (const float*)d_in[18];
  const float* b_r1     = (const float*)d_in[19];
  const float* w_r2     = (const float*)d_in[20];
  const float* b_r2     = (const float*)d_in[21];
  const float* ln_o_g   = (const float*)d_in[22];
  const float* ln_o_b   = (const float*)d_in[23];
  const float* w_out    = (const float*)d_in[24];
  const float* b_out    = (const float*)d_in[25];

  char* base = (char*)d_ws;
  size_t off = 0;
  auto alloc = [&](size_t bytes) -> void* {
    void* p = base + off;
    off += (bytes + 255) & ~(size_t)255;
    return p;
  };

  h16* xh      = (h16*)alloc((size_t)MM * DD * 2);
  h16* gih     = (h16*)alloc((size_t)MM * 512 * 2);
  h16* wt_pv   = (h16*)alloc(256 * 256 * 2);
  h16* wt_kv   = (h16*)alloc(256 * 256 * 2);
  h16* wt_key  = (h16*)alloc(32 * 256 * 2);
  h16* wt_vg1  = (h16*)alloc(256 * 512 * 2);
  h16* wt_bg1  = (h16*)alloc(128 * 256 * 2);
  h16* wt_r1   = (h16*)alloc(512 * 256 * 2);
  h16* wt_r2   = (h16*)alloc(256 * 512 * 2);
  h16* wt_out  = (h16*)alloc(256 * 256 * 2);
  h16* vt_pos  = (h16*)alloc((size_t)BB * DD * LL * 2);
  h16* vt_kv   = (h16*)alloc((size_t)BB * DD * LL * 2);
  float* keyph = (float*)alloc((size_t)MM * PP * 4);
  h16* hh      = (h16*)alloc((size_t)MM * DD * 2);
  h16* bgh     = (h16*)alloc((size_t)MM * 128 * 2);
  h16* qpos    = (h16*)alloc((size_t)LL * 64 * 2);
  h16* qkv     = (h16*)alloc((size_t)MM * 64 * 2);
  h16* kkv     = (h16*)alloc((size_t)MM * 64 * 2);
  float* vgate = (float*)alloc((size_t)MM * 4);
  float* rscal = (float*)alloc((size_t)MM * 4);
  float* blend = (float*)alloc((size_t)MM * 2 * 4);
  float* p_ret = (float*)alloc((size_t)MM * DD * 4);
  float* k_ret = (float*)alloc((size_t)MM * DD * 4);
  float* blnd  = (float*)alloc((size_t)MM * DD * 4);
  h16* blnh    = (h16*)alloc((size_t)MM * DD * 2);
  h16* r1h     = (h16*)alloc((size_t)MM * 512 * 2);
  float* refnd = (float*)alloc((size_t)MM * DD * 4);
  h16* rlnh    = (h16*)alloc((size_t)MM * DD * 2);

  auto tw = [&](const float* w, h16* wt, int K, int N) {
    k_transpose_w<<<(K * N + 255) / 256, 256, 0, stream>>>(w, wt, K, N);
  };
  tw(w_pv, wt_pv, 256, 256);
  tw(w_kv, wt_kv, 256, 256);
  tw(w_key, wt_key, 256, 32);
  tw(w_vg1, wt_vg1, 512, 256);
  tw(w_bg1, wt_bg1, 256, 128);
  tw(w_r1, wt_r1, 256, 512);
  tw(w_r2, wt_r2, 512, 256);
  tw(w_out, wt_out, 256, 256);

  k_build_x<<<MM * DD / 256, 256, 0, stream>>>(x, xh, gih);
  k_qpos<<<LL * PP / 256, 256, 0, stream>>>(pos_ph, qpos);

  // projections (register-blocked WMMA GEMMs; grids = (M/16)*(N/64)/8 blocks)
  k_gemm4<<<128, 256, 0, stream>>>(xh, wt_pv, b_pv, vt_pos, nullptr, MM, 256, 256, 3);
  k_gemm4<<<128, 256, 0, stream>>>(xh, wt_kv, b_kv, vt_kv, nullptr, MM, 256, 256, 3);
  k_gemm<<<64, 256, 0, stream>>>(xh, wt_key, b_key, keyph, nullptr, MM, 32, 256, 4);
  k_gemm4<<<128, 256, 0, stream>>>(gih, wt_vg1, b_vg1, hh, nullptr, MM, 256, 512, 1);

  k_vg<<<MM / 8, 256, 0, stream>>>(hh, w_vg2, b_vg2, vgate);
  k_scan<<<BB, 256, 0, stream>>>(vgate, rscal);
  k_qkkv<<<MM * PP / 256, 256, 0, stream>>>(keyph, vgate, qkv, kkv);

  k_gemm4<<<64, 256, 0, stream>>>(xh, wt_bg1, b_bg1, bgh, nullptr, MM, 128, 256, 1);
  k_blend<<<MM / 8, 256, 0, stream>>>(bgh, w_bg2, b_bg2, blend);

  // causal linear attentions (the big WMMA work, async LDS staging)
  k_attn<<<dim3(LL / 64, BB), 256, 0, stream>>>(qpos, qpos, vt_pos, nullptr,
                                                0.17677669529663689f, p_ret, 0);
  k_attn<<<dim3(LL / 64, BB), 256, 0, stream>>>(qkv, kkv, vt_kv, rscal, 0.0f, k_ret,
                                                LL * 64);

  k_combine<<<MM * DD / 256, 256, 0, stream>>>(p_ret, k_ret, blend, blnd);
  k_ln<<<MM, 256, 0, stream>>>(blnd, ln_r_g, ln_r_b, blnh);
  k_gemm4<<<256, 256, 0, stream>>>(blnh, wt_r1, b_r1, r1h, nullptr, MM, 512, 256, 1);
  k_gemm4<<<128, 256, 0, stream>>>(r1h, wt_r2, b_r2, refnd, nullptr, MM, 256, 512, 5);
  k_ln<<<MM, 256, 0, stream>>>(refnd, ln_o_g, ln_o_b, rlnh);
  k_gemm4<<<128, 256, 0, stream>>>(rlnh, wt_out, b_out, (float*)d_out, x, MM, 256, 256, 6);
}
